// VMDWithZHRec_46291157516343
// MI455X (gfx1250) — matrix operand order repaired and checked by reading, not compiled, not run on hardware
//
#include <hip/hip_runtime.h>

// MI455X / gfx1250, wave32. All GEMMs via v_wmma_f32_16x16x32_bf16 with
// pre-converted bf16 weights staged in d_ws (one-shot repack kernel).
// M-tile = 32 rows/workgroup: each B (weight) fragment is reused by two
// WMMAs (two M halves) to halve L2 weight traffic.

typedef __attribute__((ext_vector_type(16))) __bf16 v16bf;
typedef __attribute__((ext_vector_type(8)))  __bf16 v8bf;
typedef __attribute__((ext_vector_type(8)))  float  v8f;

#define Bb   4096
#define Tt   32
#define Xx   256
#define Hh   256
#define Zz   128
#define Yy   2
#define Gg   128
#define OW   (Gg + Yy + Tt)   // 162
#define CATW 640              // [x(256)|h(256)|z(128)]
#define MT   32               // batch rows per workgroup

// bf16 weight workspace layout (element offsets)
#define OFF_WIH 0u            // [768][256]
#define OFF_WHH 196608u       // [768][256]
#define OFF_WPH 393216u       // [128][640]
#define OFF_WOH 475136u       // [128][640]  (x|h|z cols; y cols dropped)
#define OFF_WPM 557056u       // [128][128]
#define OFF_WPS 573440u
#define OFF_WQM 589824u
#define OFF_WQS 606208u
#define OFF_WG  622592u       // [128][384]
#define W_TOTAL 671744u

// ---- one-shot fp32 -> bf16 weight repack into workspace ----
__global__ void cvt_weights_kernel(const float* __restrict__ W_ih,
                                   const float* __restrict__ W_hh,
                                   const float* __restrict__ Wph,
                                   const float* __restrict__ Woh,
                                   const float* __restrict__ Wpm,
                                   const float* __restrict__ Wps,
                                   const float* __restrict__ Wqm,
                                   const float* __restrict__ Wqs,
                                   const float* __restrict__ Wg,
                                   __bf16* __restrict__ ws) {
  unsigned i = blockIdx.x * blockDim.x + threadIdx.x;
  if (i >= W_TOTAL) return;
  float v;
  if (i < OFF_WHH)      v = W_ih[i - OFF_WIH];
  else if (i < OFF_WPH) v = W_hh[i - OFF_WHH];
  else if (i < OFF_WOH) v = Wph[i - OFF_WPH];
  else if (i < OFF_WPM) {
    unsigned j = i - OFF_WOH, r = j / 640u, c = j % 640u;
    v = Woh[(size_t)r * 642u + (c < 512u ? c : c + 2u)];  // skip y cols
  } else if (i < OFF_WPS) v = Wpm[i - OFF_WPM];
  else if (i < OFF_WQM)   v = Wps[i - OFF_WPS];
  else if (i < OFF_WQS)   v = Wqm[i - OFF_WQM];
  else if (i < OFF_WG)    v = Wqs[i - OFF_WQS];
  else                    v = Wg[i - OFF_WG];
  ws[i] = (__bf16)v;
}

#define SHUF16(lo, hi)                                                        \
  __builtin_shufflevector(lo, hi, 0, 1, 2, 3, 4, 5, 6, 7, 8, 9, 10, 11, 12,  \
                          13, 14, 15)

// Two 16x16 D tiles (rows 0-15 and 16-31 of a 32-row A panel) sharing one
// B fragment per K-chunk: B loaded once from global, used by two WMMAs.
__device__ __forceinline__ void gemm2_bf16(v8f& acc0, v8f& acc1,
                                           const __bf16* A0, int lda,
                                           const __bf16* W0, int ldw, int K,
                                           int lane) {
  const int ln = lane & 15;
  const int ka = (lane >> 4) << 3;   // A: K halves 0..7/16..23 vs 8..15/24..31
  const int kb = (lane >> 4) << 3;   // B: K halves 0..7/16..23 vs 8..15/24..31
  const __bf16* ap0 = A0 + ln * lda + ka;
  const __bf16* ap1 = ap0 + 16 * lda;
  const __bf16* wp  = W0 + (size_t)ln * ldw + kb;
  for (int k0 = 0; k0 < K; k0 += 32) {
    v8bf b0 = *(const v8bf*)(wp + k0);
    v8bf b1 = *(const v8bf*)(wp + k0 + 16);
    v16bf b = SHUF16(b0, b1);
    v8bf a00 = *(const v8bf*)(ap0 + k0);
    v8bf a01 = *(const v8bf*)(ap0 + k0 + 16);
    v8bf a10 = *(const v8bf*)(ap1 + k0);
    v8bf a11 = *(const v8bf*)(ap1 + k0 + 16);
    acc0 = __builtin_amdgcn_wmma_f32_16x16x32_bf16(
        false, SHUF16(a00, a01), false, b, (short)0, acc0, false, false);
    acc1 = __builtin_amdgcn_wmma_f32_16x16x32_bf16(
        false, SHUF16(a10, a11), false, b, (short)0, acc1, false, false);
  }
}

__global__ __launch_bounds__(256, 1) void vmd_zh_rec_kernel(
    const float* __restrict__ x, const float* __restrict__ y_true,
    const int* __restrict__ n_days, const float* __restrict__ z0,
    const float* __restrict__ eps, const __bf16* __restrict__ wsb,
    const float* __restrict__ Woh,  // fp32, only y columns 512/513 used
    const float* __restrict__ b_ih, const float* __restrict__ b_hh,
    const float* __restrict__ bph, const float* __restrict__ boh,
    const float* __restrict__ bpm, const float* __restrict__ bps,
    const float* __restrict__ bqm, const float* __restrict__ bqs,
    const float* __restrict__ bg, const float* __restrict__ Wy,
    const float* __restrict__ by, float* __restrict__ out) {
  // ---- LDS layout (phase-aliased, total 123264 B; 2 WGs fit in 320KB WGP) --
  __shared__ __align__(16) unsigned char smem[123264];
  __bf16* catb = (__bf16*)smem;             // [32][640] bf16 : x|h(masked)|z
  __bf16* hb   = (__bf16*)(smem + 40960);   // [32][256] bf16 : h carry
  unsigned char* S = smem + 57344;          // 64 KB phase scratch
  float* yrow = (float*)(smem + 122880);    // [32][2]
  int*   ndl  = (int*)(smem + 123136);      // [32]

  const int tid  = threadIdx.x;
  const int lane = tid & 31;
  const int wv   = tid >> 5;          // 8 waves
  const int ln   = lane & 15;         // output column within tile
  const int mb   = (lane >> 4) << 3;  // output row base (0 or 8)
  const int b0   = blockIdx.x * MT;

  // ---- init state ----
  for (int i = tid; i < MT * 256; i += 256) {
    hb[i] = (__bf16)0.f;
    catb[(i >> 8) * CATW + 256 + (i & 255)] = (__bf16)0.f;
  }
  for (int i = tid; i < MT * Zz; i += 256) {
    int r = i >> 7, c = i & 127;
    catb[r * CATW + 512 + c] = (__bf16)z0[(size_t)(b0 + r) * Zz + c];
  }
  if (tid < MT) ndl[tid] = n_days[b0 + tid];
  __syncthreads();

  for (int t = 0; t < Tt; ++t) {
    // ---- stage x_t (float4 loads -> bf16) and y_t ----
    for (int i = tid; i < MT * (Xx / 4); i += 256) {
      int r = i >> 6, c4 = (i & 63) << 2;
      const float4 v = ((const float4*)(x + ((size_t)(b0 + r) * Tt + t) * Xx +
                                        c4))[0];
      __bf16* d = &catb[r * CATW + c4];
      d[0] = (__bf16)v.x; d[1] = (__bf16)v.y;
      d[2] = (__bf16)v.z; d[3] = (__bf16)v.w;
    }
    if (tid < MT * 2) {
      int r = tid >> 1, c = tid & 1;
      yrow[r * 2 + c] = y_true[((size_t)(b0 + r) * Tt + t) * Yy + c];
    }
    // prefetch next timestep's x panel (global_prefetch_b8)
    if (t + 1 < Tt && tid < 128) {
      int r = tid >> 2, o = (tid & 3) << 6;
      __builtin_prefetch(x + ((size_t)(b0 + r) * Tt + (t + 1)) * Xx + o, 0, 3);
    }
    __syncthreads();

    // ---- GRU gates: 48 N-tiles of 768 cols across 8 waves ----
    __bf16* RU  = (__bf16*)S;            // [32][512] sigmoid(r|u)
    __bf16* INN = (__bf16*)(S + 32768);  // [32][256] x-part of n gate
    __bf16* HN  = (__bf16*)(S + 49152);  // [32][256] h-part of n gate
    for (int nt = wv; nt < 48; nt += 8) {
      v8f a0 = {}, a1 = {};
      gemm2_bf16(a0, a1, catb, CATW, wsb + OFF_WIH + (size_t)nt * 16 * Xx, Xx,
                 Xx, lane);
      int ng = nt * 16 + ln;
      if (nt < 32) {  // r and u gates: fuse both GEMMs into same accumulators
        gemm2_bf16(a0, a1, hb, Hh, wsb + OFF_WHH + (size_t)nt * 16 * Hh, Hh,
                   Hh, lane);
        float bia = b_ih[ng] + b_hh[ng];
#pragma unroll
        for (int rr = 0; rr < 8; ++rr) {
          RU[(mb + rr) * 512 + ng] =
              (__bf16)(1.f / (1.f + expf(-(a0[rr] + bia))));
          RU[(16 + mb + rr) * 512 + ng] =
              (__bf16)(1.f / (1.f + expf(-(a1[rr] + bia))));
        }
      } else {  // n gate: keep x- and h-parts separate
        v8f h0 = {}, h1 = {};
        gemm2_bf16(h0, h1, hb, Hh, wsb + OFF_WHH + (size_t)nt * 16 * Hh, Hh,
                   Hh, lane);
        int nc = (nt - 32) * 16 + ln;
        float bi = b_ih[ng], bh = b_hh[ng];
#pragma unroll
        for (int rr = 0; rr < 8; ++rr) {
          INN[(mb + rr) * 256 + nc]      = (__bf16)(a0[rr] + bi);
          INN[(16 + mb + rr) * 256 + nc] = (__bf16)(a1[rr] + bi);
          HN[(mb + rr) * 256 + nc]       = (__bf16)(h0[rr] + bh);
          HN[(16 + mb + rr) * 256 + nc]  = (__bf16)(h1[rr] + bh);
        }
      }
    }
    __syncthreads();

    // ---- gate combine + h update (hs masked into catb.h) ----
    for (int i = tid; i < MT * Hh; i += 256) {
      int r = i >> 8, c = i & 255;
      float rg = (float)RU[r * 512 + c];
      float ug = (float)RU[r * 512 + 256 + c];
      float nn = tanhf((float)INN[r * 256 + c] + rg * (float)HN[r * 256 + c]);
      float hnew = (1.f - ug) * nn + ug * (float)hb[r * 256 + c];
      hb[r * 256 + c] = (__bf16)hnew;  // carry (unmasked)
      catb[r * CATW + 256 + c] = (t < ndl[r]) ? (__bf16)hnew : (__bf16)0.f;
    }
    __syncthreads();

    // ---- VAE scratch views ----
    __bf16* HZP = (__bf16*)S;            // [32][128]
    __bf16* HZQ = (__bf16*)(S + 8192);   // [32][128]
    __bf16* MPc = (__bf16*)(S + 16384);  // [32][128]
    __bf16* VPc = (__bf16*)(S + 24576);  // [32][128]
    __bf16* MQc = (__bf16*)(S + 32768);  // [32][128]
    __bf16* VQc = (__bf16*)(S + 40960);  // [32][128]
    float*  KLt = (float*)(S + 49152);   // [32][128]

    // ---- prior + posterior hidden: both single K=640 GEMMs (1 tile/wave) --
    {
      int nt = wv;  // 8 tiles, 8 waves
      v8f p0 = {}, p1 = {};
      gemm2_bf16(p0, p1, catb, CATW, wsb + OFF_WPH + (size_t)nt * 16 * CATW,
                 CATW, CATW, lane);
      float bi = bph[nt * 16 + ln];
#pragma unroll
      for (int rr = 0; rr < 8; ++rr) {
        HZP[(mb + rr) * 128 + nt * 16 + ln]      = (__bf16)tanhf(p0[rr] + bi);
        HZP[(16 + mb + rr) * 128 + nt * 16 + ln] = (__bf16)tanhf(p1[rr] + bi);
      }

      v8f q0 = {}, q1 = {};
      gemm2_bf16(q0, q1, catb, CATW, wsb + OFF_WOH + (size_t)nt * 16 * CATW,
                 CATW, CATW, lane);
      float wy0 = Woh[(size_t)(nt * 16 + ln) * 642 + 512];
      float wy1 = Woh[(size_t)(nt * 16 + ln) * 642 + 513];
      float bo  = boh[nt * 16 + ln];
#pragma unroll
      for (int rr = 0; rr < 8; ++rr) {
        float v0 = q0[rr] + yrow[(mb + rr) * 2] * wy0 +
                   yrow[(mb + rr) * 2 + 1] * wy1 + bo;
        float v1 = q1[rr] + yrow[(16 + mb + rr) * 2] * wy0 +
                   yrow[(16 + mb + rr) * 2 + 1] * wy1 + bo;
        HZQ[(mb + rr) * 128 + nt * 16 + ln]      = (__bf16)tanhf(v0);
        HZQ[(16 + mb + rr) * 128 + nt * 16 + ln] = (__bf16)tanhf(v1);
      }
    }
    __syncthreads();

    // ---- mp/vp/mq/vq (four 128x128 GEMMs; 1 N-tile per wave) ----
    {
      int nt = wv;
      int ng = nt * 16 + ln;
      float c1 = bpm[ng], c2 = bps[ng], c3 = bqm[ng], c4 = bqs[ng];
      v8f am0 = {}, am1 = {}, av0 = {}, av1 = {};
      gemm2_bf16(am0, am1, HZP, Zz, wsb + OFF_WPM + (size_t)nt * 16 * Zz, Zz,
                 Zz, lane);
      gemm2_bf16(av0, av1, HZP, Zz, wsb + OFF_WPS + (size_t)nt * 16 * Zz, Zz,
                 Zz, lane);
#pragma unroll
      for (int rr = 0; rr < 8; ++rr) {
        MPc[(mb + rr) * 128 + ng]      = (__bf16)(am0[rr] + c1);
        MPc[(16 + mb + rr) * 128 + ng] = (__bf16)(am1[rr] + c1);
        VPc[(mb + rr) * 128 + ng]      = (__bf16)(av0[rr] + c2);
        VPc[(16 + mb + rr) * 128 + ng] = (__bf16)(av1[rr] + c2);
      }
      v8f bm0 = {}, bm1 = {}, bv0 = {}, bv1 = {};
      gemm2_bf16(bm0, bm1, HZQ, Zz, wsb + OFF_WQM + (size_t)nt * 16 * Zz, Zz,
                 Zz, lane);
      gemm2_bf16(bv0, bv1, HZQ, Zz, wsb + OFF_WQS + (size_t)nt * 16 * Zz, Zz,
                 Zz, lane);
#pragma unroll
      for (int rr = 0; rr < 8; ++rr) {
        MQc[(mb + rr) * 128 + ng]      = (__bf16)(bm0[rr] + c3);
        MQc[(16 + mb + rr) * 128 + ng] = (__bf16)(bm1[rr] + c3);
        VQc[(mb + rr) * 128 + ng]      = (__bf16)(bv0[rr] + c4);
        VQc[(16 + mb + rr) * 128 + ng] = (__bf16)(bv1[rr] + c4);
      }
    }
    __syncthreads();

    // ---- z_post = mq + sq*eps ; KL elementwise ----
    for (int i = tid; i < MT * Zz; i += 256) {
      int r = i >> 7, c = i & 127;
      float mpv = (float)MPc[r * 128 + c], vpv = (float)VPc[r * 128 + c];
      float mqv = (float)MQc[r * 128 + c], vqv = (float)VQc[r * 128 + c];
      float sp = expf(0.5f * vpv), sq = expf(0.5f * vqv);
      float e  = eps[((size_t)t * Bb + (b0 + r)) * Zz + c];
      catb[r * CATW + 512 + c] = (__bf16)(mqv + sq * e);
      float d = mqv - mpv;
      KLt[r * 128 + c] =
          0.5f * (vpv - vqv) + (sq * sq + d * d) / (2.f * sp * sp) - 0.5f;
    }
    __syncthreads();

    // ---- g = tanh([hs|z] @ Wg^T + bg), K=384 (1 N-tile per wave) ----
    float* GF = (float*)S;  // [32][128] fp32, overlaps HZP..VQc (dead now)
    {
      int nt = wv;
      v8f g0 = {}, g1 = {};
      gemm2_bf16(g0, g1, catb + 256, CATW, wsb + OFF_WG + (size_t)nt * 16 * 384,
                 384, 384, lane);
      float bi = bg[nt * 16 + ln];
#pragma unroll
      for (int rr = 0; rr < 8; ++rr) {
        GF[(mb + rr) * 128 + nt * 16 + ln]      = tanhf(g0[rr] + bi);
        GF[(16 + mb + rr) * 128 + nt * 16 + ln] = tanhf(g1[rr] + bi);
      }
    }
    __syncthreads();

    // ---- KL row-sum, Y=2 softmax head, last-day select ----
    if (tid < MT) {
      int r = tid, b = b0 + r;
      float kls = 0.f;
      for (int c = 0; c < 128; ++c) kls += KLt[r * 128 + c];
      out[(size_t)b * OW + Gg + Yy + t] = kls;
      float l0 = by[0], l1 = by[1];
      for (int c = 0; c < 128; ++c) {
        float gv = GF[r * 128 + c];
        l0 += gv * Wy[c];
        l1 += gv * Wy[128 + c];
      }
      if (t == ndl[r] - 1) {
        float mx = fmaxf(l0, l1);
        float e0 = expf(l0 - mx), e1 = expf(l1 - mx), si = 1.f / (e0 + e1);
        float* orow = out + (size_t)b * OW;
        for (int c = 0; c < 128; ++c) orow[c] = GF[r * 128 + c];
        orow[128] = e0 * si;
        orow[129] = e1 * si;
      }
    }
    __syncthreads();
  }
}

extern "C" void kernel_launch(void* const* d_in, const int* in_sizes, int n_in,
                              void* d_out, int out_size, void* d_ws,
                              size_t ws_size, hipStream_t stream) {
  (void)in_sizes; (void)n_in; (void)out_size; (void)ws_size;
  const float* x      = (const float*)d_in[0];
  const float* y_true = (const float*)d_in[1];
  const int*   n_days = (const int*)d_in[2];
  const float* z0     = (const float*)d_in[3];
  const float* eps    = (const float*)d_in[4];
  const float* W_ih   = (const float*)d_in[5];
  const float* W_hh   = (const float*)d_in[6];
  const float* b_ih   = (const float*)d_in[7];
  const float* b_hh   = (const float*)d_in[8];
  const float* Wph    = (const float*)d_in[9];
  const float* bph    = (const float*)d_in[10];
  const float* Woh    = (const float*)d_in[11];
  const float* boh    = (const float*)d_in[12];
  const float* Wpm    = (const float*)d_in[13];
  const float* bpm    = (const float*)d_in[14];
  const float* Wps    = (const float*)d_in[15];
  const float* bps    = (const float*)d_in[16];
  const float* Wqm    = (const float*)d_in[17];
  const float* bqm    = (const float*)d_in[18];
  const float* Wqs    = (const float*)d_in[19];
  const float* bqs    = (const float*)d_in[20];
  const float* Wg     = (const float*)d_in[21];
  const float* bg     = (const float*)d_in[22];
  const float* Wy     = (const float*)d_in[23];
  const float* by     = (const float*)d_in[24];

  __bf16* wsb = (__bf16*)d_ws;  // ~1.31 MB of bf16 weights

  // 1) one-shot weight repack (stream-ordered before the main kernel)
  cvt_weights_kernel<<<(W_TOTAL + 255) / 256, 256, 0, stream>>>(
      W_ih, W_hh, Wph, Woh, Wpm, Wps, Wqm, Wqs, Wg, wsb);

  // 2) fused recurrent VAE: 128 workgroups x 8 waves, 32 batch rows each
  dim3 grid(Bb / MT), block(256);
  vmd_zh_rec_kernel<<<grid, block, 0, stream>>>(
      x, y_true, n_days, z0, eps, wsb, Woh, b_ih, b_hh, bph, boh, bpm, bps,
      bqm, bqs, bg, Wy, by, (float*)d_out);
}
